// SelfAttention_67396626809397
// MI455X (gfx1250) — matrix operand order repaired and checked by reading, not compiled, not run on hardware
//
#include <hip/hip_runtime.h>

typedef __attribute__((ext_vector_type(16))) _Float16 v16h;
typedef __attribute__((ext_vector_type(8)))  _Float16 half8;
typedef __attribute__((ext_vector_type(8)))  float    v8f;

#define BB   4
#define CCH  64
#define DQK  8
#define NPOS 4096   // 64*64
#define LOG2E 1.4426950408889634f

union V16 { v16h v; half8 h[2]; };

// ---- DPP16 all-reduce across each 16-lane row (wave32 halves stay separate) ----
template <int CTRL>
__device__ __forceinline__ float dpp_movf(float v) {
    return __builtin_bit_cast(float,
        __builtin_amdgcn_update_dpp(0, __builtin_bit_cast(int, v),
                                    CTRL, 0xF, 0xF, true));
}
__device__ __forceinline__ float row16_max(float v) {
    v = fmaxf(v, dpp_movf<0xB1>(v));   // quad_perm(1,0,3,2)  : xor 1
    v = fmaxf(v, dpp_movf<0x4E>(v));   // quad_perm(2,3,0,1)  : xor 2
    v = fmaxf(v, dpp_movf<0x141>(v));  // row_half_mirror     : fold 4<->4
    v = fmaxf(v, dpp_movf<0x140>(v));  // row_mirror          : fold 8<->8
    return v;
}
__device__ __forceinline__ float row16_sum(float v) {
    v += dpp_movf<0xB1>(v);
    v += dpp_movf<0x4E>(v);
    v += dpp_movf<0x141>(v);
    v += dpp_movf<0x140>(v);
    return v;
}

// ---------------------------------------------------------------------------
// Projection: q = (Wq x + bq)*log2(e) -> f16 [B][N][8]  (base-2 softmax fold)
//             k =  Wk x + bk          -> f16 [B][N][8]
//             v =  Wv x + bv          -> f16 [B][C][N'] where each 64-key block
//                  is stored key-permuted: j = (n%16)*4 + n/16 (matches P order)
// ---------------------------------------------------------------------------
__global__ __launch_bounds__(256) void proj_kernel(
    const float* __restrict__ x,
    const float* __restrict__ Wq, const float* __restrict__ bq,
    const float* __restrict__ Wk, const float* __restrict__ bk,
    const float* __restrict__ Wv, const float* __restrict__ bv,
    _Float16* __restrict__ q16, _Float16* __restrict__ k16,
    _Float16* __restrict__ vT16)
{
    __shared__ float xs[64][66];
    __shared__ float wv[64][65];
    __shared__ float wq[8][65];
    __shared__ float wk[8][65];

    const int t   = threadIdx.x;
    const int blk = blockIdx.x;          // 0..255
    const int b   = blk >> 6;
    const int n0  = (blk & 63) * 64;

    const float* xb = x + (size_t)b * CCH * NPOS;

    {   // stage x tile, coalesced along n
        const int n_l = t & 63;
        const int c0  = t >> 6;
        for (int i = 0; i < 16; ++i) {
            const int c = c0 + i * 4;
            xs[c][n_l] = xb[(size_t)c * NPOS + n0 + n_l];
        }
    }
    for (int i = 0; i < 16; ++i) {
        const int idx = t + i * 256;
        wv[idx >> 6][idx & 63] = Wv[idx];
    }
    for (int i = 0; i < 2; ++i) {
        const int idx = t + i * 256;
        wq[idx >> 6][idx & 63] = Wq[idx];
        wk[idx >> 6][idx & 63] = Wk[idx];
    }
    __syncthreads();

    {   // V: channel e for 16 positions n_local = ng*16 + j (j = 0..15)
        const int e  = t & 63;
        const int ng = t >> 6;
        float acc[16];
        const float bve = bv[e];
        for (int j = 0; j < 16; ++j) acc[j] = bve;
        for (int c = 0; c < 64; ++c) {
            const float w = wv[e][c];
            for (int j = 0; j < 16; ++j) acc[j] += w * xs[c][ng * 16 + j];
        }
        // permuted store within the 64-key block: position = j*4 + ng
        _Float16* vp = vT16 + ((size_t)b * CCH + e) * NPOS + n0;
        for (int j = 0; j < 16; ++j) vp[j * 4 + ng] = (_Float16)acc[j];
    }

    {   // Q/K: threads 0..127 -> q (scaled by log2e), 128..255 -> k
        const int  d    = t & 7;
        const int  nl0  = ((t >> 3) & 15) * 4;
        const bool isK  = (t >= 128);
        const float (*wp)[65] = isK ? wk : wq;
        const float bias = isK ? bk[d] : bq[d];
        const float post = isK ? 1.0f : LOG2E;
        float acc[4];
        for (int j = 0; j < 4; ++j) acc[j] = bias;
        for (int c = 0; c < 64; ++c) {
            const float w = wp[d][c];
            for (int j = 0; j < 4; ++j) acc[j] += w * xs[c][nl0 + j];
        }
        _Float16* outp = isK ? k16 : q16;
        for (int j = 0; j < 4; ++j)
            outp[((size_t)b * NPOS + n0 + nl0 + j) * DQK + d] = (_Float16)(acc[j] * post);
    }
}

// ---------------------------------------------------------------------------
// Flash attention: block = 64 queries (4 waves x 16), keys in chunks of 64.
// Per chunk: 4 score WMMAs, DPP online-softmax, P packed to f16 LDS with
// v_cvt_pk (key order permuted to make each lane's 4 probs adjacent),
// 8 P*V WMMAs against the identically-permuted V.
// ---------------------------------------------------------------------------
__global__ __launch_bounds__(128) void attn_kernel(
    const _Float16* __restrict__ q16, const _Float16* __restrict__ k16,
    const _Float16* __restrict__ vT16, const float* __restrict__ x,
    const float* __restrict__ gamma_p, float* __restrict__ out)
{
    __shared__ __align__(16) _Float16 plds[4][16][72];  // [wave][row][j], 64+8 pad

    const int lane = threadIdx.x & 31;
    const int wave = threadIdx.x >> 5;
    const int hi   = lane >> 4;
    const int lo   = lane & 15;
    const int blk  = blockIdx.x;
    const int b    = blk >> 6;
    const int n0   = (blk & 63) * 64 + wave * 16;

    // Q A-operand: M=16 queries, reduction K=32 (d 0..7 real, rest zero).
    V16 qa; qa.v = v16h{};
    if (hi == 0)
        qa.h[0] = *(const half8*)(q16 + ((size_t)b * NPOS + n0 + lo) * DQK);

    v8f acc[4];
    const v8f zc = v8f{};
    for (int ct = 0; ct < 4; ++ct) acc[ct] = zc;
    float mrow[8], lrow[8];
    for (int r = 0; r < 8; ++r) { mrow[r] = -1e30f; lrow[r] = 0.0f; }

    const _Float16* kb_base = k16  + (size_t)b * NPOS * DQK;
    const _Float16* vb_base = vT16 + (size_t)b * CCH * NPOS;

    for (int m0 = 0; m0 < NPOS; m0 += 64) {
        // ---- scores: 4 key tiles of 16 (natural key order) ----
        V16 kb[4];
        for (int kt = 0; kt < 4; ++kt) {
            kb[kt].v = v16h{};
            if (hi == 0)
                kb[kt].h[0] = *(const half8*)(kb_base + (size_t)(m0 + kt * 16 + lo) * DQK);
        }
        if (m0 + 64 < NPOS) {
            __builtin_prefetch(kb_base + (size_t)(m0 + 64 + lane) * DQK, 0, 3);
            __builtin_prefetch(kb_base + (size_t)(m0 + 96 + lane) * DQK, 0, 3);
            __builtin_prefetch(vb_base + (size_t)lane * NPOS + m0 + 64, 0, 3);
        }
        v8f s[4];
        for (int kt = 0; kt < 4; ++kt)
            s[kt] = __builtin_amdgcn_wmma_f32_16x16x32_f16(false, qa.v, false, kb[kt].v,
                                                           (short)0, zc, false, false);

        // ---- online softmax (base-2; row M = r + 8*hi lives in this half) ----
        float tmax[8];
        for (int r = 0; r < 8; ++r)
            tmax[r] = fmaxf(fmaxf(s[0][r], s[1][r]), fmaxf(s[2][r], s[3][r]));
        for (int r = 0; r < 8; ++r) tmax[r] = row16_max(tmax[r]);

        float p[4][8], rs[8];
        for (int r = 0; r < 8; ++r) {
            const float mn = fmaxf(mrow[r], tmax[r]);
            const float sc = __builtin_exp2f(mrow[r] - mn);
            mrow[r]  = mn;
            lrow[r] *= sc;
            for (int ct = 0; ct < 4; ++ct) acc[ct][r] *= sc;
            for (int kt = 0; kt < 4; ++kt) p[kt][r] = __builtin_exp2f(s[kt][r] - mn);
            rs[r] = (p[0][r] + p[1][r]) + (p[2][r] + p[3][r]);
        }
        for (int r = 0; r < 8; ++r) lrow[r] += row16_sum(rs[r]);

        // ---- P -> f16 LDS, permuted key order j = lo*4 + kt:
        //      lane's 4 probs per row are adjacent -> packed b64 stores ----
        for (int r = 0; r < 8; ++r) {
            const int qr = r + 8 * hi;
            union { unsigned long long u64; unsigned int u32[2]; } pk;
            pk.u32[0] = __builtin_bit_cast(unsigned int,
                            __builtin_amdgcn_cvt_pkrtz(p[0][r], p[1][r]));
            pk.u32[1] = __builtin_bit_cast(unsigned int,
                            __builtin_amdgcn_cvt_pkrtz(p[2][r], p[3][r]));
            *(unsigned long long*)&plds[wave][qr][lo * 4] = pk.u64;
        }
        asm volatile("s_wait_dscnt 0x0" ::: "memory");   // cross-lane LDS RAW in-wave
        // A-operand reads: K-runs are contiguous in j by construction
        V16 pa0, pa1;
        pa0.h[0] = *(const half8*)(&plds[wave][lo][hi * 8]);
        pa0.h[1] = *(const half8*)(&plds[wave][lo][16 + hi * 8]);
        pa1.h[0] = *(const half8*)(&plds[wave][lo][32 + hi * 8]);
        pa1.h[1] = *(const half8*)(&plds[wave][lo][48 + hi * 8]);

        // ---- P*V over 64 channels; V stored in the same permuted key order ----
        for (int ct = 0; ct < 4; ++ct) {
            const _Float16* vp = vb_base + (size_t)(ct * 16 + lo) * NPOS + m0 + hi * 8;
            V16 vb0, vb1;
            vb0.h[0] = *(const half8*)(vp);
            vb0.h[1] = *(const half8*)(vp + 16);
            vb1.h[0] = *(const half8*)(vp + 32);
            vb1.h[1] = *(const half8*)(vp + 48);
            acc[ct] = __builtin_amdgcn_wmma_f32_16x16x32_f16(false, pa0.v, false, vb0.v,
                                                             (short)0, acc[ct], false, false);
            acc[ct] = __builtin_amdgcn_wmma_f32_16x16x32_f16(false, pa1.v, false, vb1.v,
                                                             (short)0, acc[ct], false, false);
        }
    }

    // ---- epilogue: normalize, gamma*out + x ----
    const float g = gamma_p[0];
    float linv[8];
    for (int r = 0; r < 8; ++r) linv[r] = 1.0f / lrow[r];
    const float* xb = x   + (size_t)b * CCH * NPOS;
    float*       ob = out + (size_t)b * CCH * NPOS;
    for (int ct = 0; ct < 4; ++ct) {
        const int c = ct * 16 + lo;
        for (int r = 0; r < 8; ++r) {
            const int n = n0 + r + 8 * hi;
            ob[(size_t)c * NPOS + n] = g * (acc[ct][r] * linv[r]) + xb[(size_t)c * NPOS + n];
        }
    }
}

// ---------------------------------------------------------------------------
extern "C" void kernel_launch(void* const* d_in, const int* in_sizes, int n_in,
                              void* d_out, int out_size, void* d_ws, size_t ws_size,
                              hipStream_t stream) {
    (void)in_sizes; (void)n_in; (void)out_size; (void)ws_size;
    const float* x     = (const float*)d_in[0];
    const float* Wq    = (const float*)d_in[1];
    const float* bq    = (const float*)d_in[2];
    const float* Wk    = (const float*)d_in[3];
    const float* bk    = (const float*)d_in[4];
    const float* Wv    = (const float*)d_in[5];
    const float* bv    = (const float*)d_in[6];
    const float* gamma = (const float*)d_in[7];
    float* out = (float*)d_out;

    char* ws = (char*)d_ws;
    _Float16* q16  = (_Float16*)(ws);                         // 256 KiB
    _Float16* k16  = (_Float16*)(ws + (size_t)256 * 1024);    // 256 KiB
    _Float16* vT16 = (_Float16*)(ws + (size_t)512 * 1024);    // 2 MiB

    proj_kernel<<<BB * (NPOS / 64), 256, 0, stream>>>(x, Wq, bq, Wk, bk, Wv, bv,
                                                      q16, k16, vT16);
    attn_kernel<<<BB * (NPOS / 64), 128, 0, stream>>>(q16, k16, vT16, x, gamma, out);
}
